// PointCloudsNN_91328184582475
// MI455X (gfx1250) — compile-verified, matrix operand order checked by reading, not run on hardware
//
#include <hip/hip_runtime.h>
#include <hip/hip_bf16.h>

typedef _Float16 h16v __attribute__((ext_vector_type(16)));
typedef _Float16 h8v  __attribute__((ext_vector_type(8)));
typedef float    f8v  __attribute__((ext_vector_type(8)));

#define NPTS 65536
#define WID  64

__device__ __forceinline__ float fast_tanh(float x) {
    // tanh(x) = 1 - 2/(exp(2x)+1): v_exp_f32 + v_rcp_f32, saturates correctly at +-1
    float e = __expf(2.0f * x);
    return 1.0f - 2.0f * __builtin_amdgcn_rcpf(e + 1.0f);
}

__device__ __forceinline__ h16v make_frag(const _Float16* p0, const _Float16* p1) {
    h8v lo = *(const h8v*)p0;   // ds_load_b128
    h8v hi = *(const h8v*)p1;   // ds_load_b128
    return __builtin_shufflevector(lo, hi, 0,1,2,3,4,5,6,7,8,9,10,11,12,13,14,15);
}

// One fused layer: out = tanh(in(1x64) * W^T(64x64) + b), vector broadcast to all
// 16 A-rows so every D row carries the result (no EXEC games, no extraction shuffle).
__device__ __forceinline__ void wmma_layer(const _Float16* in, _Float16* out,
                                           const _Float16* Wm, const float* Bv,
                                           int lane) {
    const int hl = (lane & 16) ? 8 : 0;   // half-wave K-offset per A/B f16 layout
    const int nl = lane & 15;
    // A frags (K=0..31 and K=32..63), identical rows on all lanes (broadcast M)
    h16v a0 = make_frag(in + hl,        in + hl + 16);
    h16v a1 = make_frag(in + 32 + hl,   in + 48 + hl);
#pragma unroll
    for (int t = 0; t < 4; ++t) {
        const _Float16* wr = Wm + (size_t)(t * 16 + nl) * 64 + hl; // row n = t*16+nl of W
        h16v b0 = make_frag(wr,      wr + 16);
        h16v b1 = make_frag(wr + 32, wr + 48);
        f8v acc = {0.f,0.f,0.f,0.f,0.f,0.f,0.f,0.f};
        acc = __builtin_amdgcn_wmma_f32_16x16x32_f16(false, a0, false, b0,
                                                     (short)0, acc, false, false);
        acc = __builtin_amdgcn_wmma_f32_16x16x32_f16(false, a1, false, b1,
                                                     (short)0, acc, false, false);
        float y = fast_tanh(acc[0] + Bv[t * 16 + nl]);
        out[t * 16 + nl] = (_Float16)y;   // lanes 16..31 duplicate same value/addr: benign
    }
}

struct EncArgs {
    const float* pc;                  // 65536 x 3
    const float* t1W[3]; const float* t1b[3];
    const float* mW[4];  const float* mB[4];   // t2, t2_2(61x64), t3, t3_2
};
struct ScanArgs { EncArgs e[2]; float* out; };

__global__ void __launch_bounds__(32)
scan_kernel(ScanArgs args) {
    __shared__ __align__(16) _Float16 wlds[4][WID * WID];  // fp16 weights, row-major
    __shared__ float blds[4][WID];
    __shared__ __align__(16) _Float16 cb[2][WID];          // carry ping-pong
    const int lane = threadIdx.x;
    const EncArgs& e = args.e[blockIdx.x];

    // ---- one-time: convert weights to fp16 in LDS (pad t2_2 to 64x64 with zeros) ----
#pragma unroll
    for (int m = 0; m < 4; ++m) {
        const int rows = (m == 1) ? 61 : 64;
        const float* src = e.mW[m];
        for (int i = lane; i < WID * WID; i += 32) {
            int r = i >> 6;
            wlds[m][i] = (_Float16)((r < rows) ? src[i] : 0.0f);
        }
        const float* bsrc = e.mB[m];
        for (int i = lane; i < WID; i += 32)
            blds[m][i] = (i < rows) ? bsrc[i] : 0.0f;
    }
    // tiny 3x3 chain weights -> registers (uniform -> s_loads)
    float t1w[3][9], t1b[3][3];
#pragma unroll
    for (int l = 0; l < 3; ++l) {
#pragma unroll
        for (int i = 0; i < 9; ++i) t1w[l][i] = e.t1W[l][i];
#pragma unroll
        for (int i = 0; i < 3; ++i) t1b[l][i] = e.t1b[l][i];
    }
    cb[0][lane] = (_Float16)0.0f;
    cb[0][lane + 32] = (_Float16)0.0f;
    __syncthreads();

    _Float16 hx = (_Float16)0.f, hy = (_Float16)0.f, hz = (_Float16)0.f;

    for (int step = 0; step < NPTS; ++step) {
        if ((step & 31) == 0) {
            // each lane owns one step of the next 32: load point + run t1 chain now
            const float* p = e.pc + (size_t)(step + lane) * 3;
            float x = p[0], y = p[1], z = p[2];
            if (step + 32 < NPTS)
                __builtin_prefetch(e.pc + (size_t)(step + 32 + lane) * 3, 0, 0);
#pragma unroll
            for (int l = 0; l < 3; ++l) {
                float nx = fast_tanh(t1w[l][0]*x + t1w[l][1]*y + t1w[l][2]*z + t1b[l][0]);
                float ny = fast_tanh(t1w[l][3]*x + t1w[l][4]*y + t1w[l][5]*z + t1b[l][1]);
                float nz = fast_tanh(t1w[l][6]*x + t1w[l][7]*y + t1w[l][8]*z + t1b[l][2]);
                x = nx; y = ny; z = nz;
            }
            hx = (_Float16)x; hy = (_Float16)y; hz = (_Float16)z;
        }
        wmma_layer(cb[0], cb[1], wlds[0], blds[0], lane);   // t2
        __syncthreads();
        wmma_layer(cb[1], cb[0], wlds[1], blds[1], lane);   // t2_2 (padded)
        if (lane == (step & 31)) {                           // inject t1_i -> concat[61..63]
            cb[0][61] = hx; cb[0][62] = hy; cb[0][63] = hz;
        }
        __syncthreads();
        wmma_layer(cb[0], cb[1], wlds[2], blds[2], lane);   // t3
        __syncthreads();
        wmma_layer(cb[1], cb[0], wlds[3], blds[3], lane);   // t3_2 -> new carry
        __syncthreads();
    }
    float* o = args.out + (size_t)blockIdx.x * WID;
    o[lane]      = (float)cb[0][lane];
    o[lane + 32] = (float)cb[0][lane + 32];
}

struct HeadArgs {
    const float* fW[4]; const float* fB[4];
    const float* rW[3]; const float* rB[3];
    const float* tW[3]; const float* tB[3];
    const float* vin;   // ws: [s(64), d(64)]
    float* out;         // 6 floats
};

__global__ void __launch_bounds__(32)
head_kernel(HeadArgs a) {
    __shared__ float v[128], t[128], rs1[64], ts1[64], rs2[5], ts2[5];
    const int lane = threadIdx.x;
#pragma unroll
    for (int i = 0; i < 4; ++i) v[lane + 32 * i] = a.vin[lane + 32 * i];
    __syncthreads();
    float* src = v; float* dst = t;
#pragma unroll
    for (int l = 0; l < 4; ++l) {
        const float* Wp = a.fW[l]; const float* Bp = a.fB[l];
        float acc[4];
#pragma unroll
        for (int j = 0; j < 4; ++j) {
            const int o = lane * 4 + j;
            float s = Bp[o];
            for (int k = 0; k < 128; ++k) s += Wp[o * 128 + k] * src[k];
            acc[j] = fast_tanh(s);
        }
        __syncthreads();
#pragma unroll
        for (int j = 0; j < 4; ++j) dst[lane * 4 + j] = acc[j];
        __syncthreads();
        float* tmp = src; src = dst; dst = tmp;
    }
    // src == v here (4 swaps). rot input = src[0..63], trans input = src[64..127]
    {
        float sr0 = a.rB[0][lane], sr1 = a.rB[0][lane + 32];
        float st0 = a.tB[0][lane], st1 = a.tB[0][lane + 32];
        for (int k = 0; k < 64; ++k) {
            sr0 += a.rW[0][lane * 64 + k]        * src[k];
            sr1 += a.rW[0][(lane + 32) * 64 + k] * src[k];
            st0 += a.tW[0][lane * 64 + k]        * src[64 + k];
            st1 += a.tW[0][(lane + 32) * 64 + k] * src[64 + k];
        }
        rs1[lane] = fast_tanh(sr0); rs1[lane + 32] = fast_tanh(sr1);
        ts1[lane] = fast_tanh(st0); ts1[lane + 32] = fast_tanh(st1);
    }
    __syncthreads();
    if (lane < 5) {
        float s = a.rB[1][lane], s2 = a.tB[1][lane];
        for (int k = 0; k < 64; ++k) {
            s  += a.rW[1][lane * 64 + k] * rs1[k];
            s2 += a.tW[1][lane * 64 + k] * ts1[k];
        }
        rs2[lane] = fast_tanh(s); ts2[lane] = fast_tanh(s2);
    }
    __syncthreads();
    if (lane < 3) {
        float s = a.rB[2][lane], s2 = a.tB[2][lane];
        for (int k = 0; k < 5; ++k) {
            s  += a.rW[2][lane * 5 + k] * rs2[k];
            s2 += a.tW[2][lane * 5 + k] * ts2[k];
        }
        a.out[lane]     = s;   // rot
        a.out[3 + lane] = s2;  // trans
    }
}

extern "C" void kernel_launch(void* const* d_in, const int* in_sizes, int n_in,
                              void* d_out, int out_size, void* d_ws, size_t ws_size,
                              hipStream_t stream) {
    (void)in_sizes; (void)n_in; (void)out_size; (void)ws_size;
    auto F = [&](int i) { return (const float*)d_in[i]; };
    // input order: src_pc, dst_pc, enc1{t1_W,t1_b,t1_2_W,t1_2_b,t1_3_W,t1_3_b,
    //   t2_W,t2_b,t2_2_W,t2_2_b,t3_W,t3_b,t3_2_W,t3_2_b}, enc2{...}, fin{...}
    ScanArgs sa;
    for (int b = 0; b < 2; ++b) {
        const int base = 2 + b * 14;
        EncArgs& e = sa.e[b];
        e.pc = F(b);
        e.t1W[0] = F(base + 0);  e.t1b[0] = F(base + 1);
        e.t1W[1] = F(base + 2);  e.t1b[1] = F(base + 3);
        e.t1W[2] = F(base + 4);  e.t1b[2] = F(base + 5);
        e.mW[0]  = F(base + 6);  e.mB[0]  = F(base + 7);    // t2
        e.mW[1]  = F(base + 8);  e.mB[1]  = F(base + 9);    // t2_2
        e.mW[2]  = F(base + 10); e.mB[2]  = F(base + 11);   // t3
        e.mW[3]  = F(base + 12); e.mB[3]  = F(base + 13);   // t3_2
    }
    float* ws = (float*)d_ws;
    sa.out = ws;                                  // ws[0..63]=s, ws[64..127]=d
    hipLaunchKernelGGL(scan_kernel, dim3(2), dim3(32), 0, stream, sa);

    HeadArgs ha;
    const int fb = 30;
    for (int l = 0; l < 4; ++l) { ha.fW[l] = F(fb + 2 * l); ha.fB[l] = F(fb + 2 * l + 1); }
    const int rb = 38;
    for (int l = 0; l < 3; ++l) { ha.rW[l] = F(rb + 2 * l); ha.rB[l] = F(rb + 2 * l + 1); }
    const int tb = 44;
    for (int l = 0; l < 3; ++l) { ha.tW[l] = F(tb + 2 * l); ha.tB[l] = F(tb + 2 * l + 1); }
    ha.vin = ws;
    ha.out = (float*)d_out;
    hipLaunchKernelGGL(head_kernel, dim3(1), dim3(32), 0, stream, ha);
}
